// GATNet_pyg_21071109554334
// MI455X (gfx1250) — matrix operand order repaired and checked by reading, not compiled, not run on hardware
//
#include <hip/hip_runtime.h>
#include <hip/hip_bf16.h>
#include <math.h>

typedef __bf16 bf16;
typedef __attribute__((ext_vector_type(16))) __bf16 v16bf;
typedef __attribute__((ext_vector_type(8)))  __bf16 v8bf;
typedef __attribute__((ext_vector_type(8)))  float   v8f;

#define NEG_SLOPE 0.2f
#define BN_EPS    1e-5f

// ---------------- utility kernels ----------------

__global__ __launch_bounds__(256) void fill_f32_kernel(float* __restrict__ p, float v, int n) {
    int t = blockIdx.x * blockDim.x + threadIdx.x;
    if (t < n) p[t] = v;
}

__global__ __launch_bounds__(256) void fill_i32_kernel(int* __restrict__ p, int v, int n) {
    int t = blockIdx.x * blockDim.x + threadIdx.x;
    if (t < n) p[t] = v;
}

__global__ __launch_bounds__(256) void f32_to_bf16_kernel(const float* __restrict__ a,
                                                          bf16* __restrict__ b, int n) {
    int t = blockIdx.x * blockDim.x + threadIdx.x;
    if (t < n) b[t] = (bf16)a[t];
}

// W is [K][Nc] row-major f32 -> Wt is [Nc][K] bf16 (B-operand friendly)
__global__ __launch_bounds__(256) void transpose_to_bf16_kernel(const float* __restrict__ W,
                                                                bf16* __restrict__ Wt,
                                                                int K, int Nc) {
    int t = blockIdx.x * blockDim.x + threadIdx.x;
    if (t >= K * Nc) return;
    int n = t / K, k = t - n * K;
    Wt[t] = (bf16)W[(size_t)k * Nc + n];
}

// ---------------- WMMA GEMM: out = X[M,K] * W[K,Nc] (+bias)(+relu) ----------------
// X: bf16 row-major. Wt: bf16 [Nc][K] (transposed weights). One wave = one 16x16 tile.
// A frag (16x32 bf16): lane m=lane&15, g=lane>>4; elem j -> K=g*16+j
// => two contiguous 16B loads (matches B order).
// B frag (32x16): lane n=lane&15, g=lane>>4; elem j -> K=g*16+j => one contiguous 32B run.
// D: VGPR r holds row r + 8*g, col = lane&15.
__global__ __launch_bounds__(256) void wmma_gemm_kernel(
    const bf16* __restrict__ X, const bf16* __restrict__ Wt,
    const float* __restrict__ bias, float* __restrict__ outF, bf16* __restrict__ outB,
    int M, int K, int Nc, int relu)
{
    int wave   = blockIdx.x * (blockDim.x >> 5) + (threadIdx.x >> 5);
    int tilesN = Nc >> 4;
    int tilesM = M >> 4;
    if (wave >= tilesM * tilesN) return;         // uniform per wave: EXEC all-ones below
    int tm = wave / tilesN, tn = wave - tm * tilesN;
    int lane = threadIdx.x & 31;
    int nn = lane & 15, g = lane >> 4;
    size_t row0 = (size_t)tm << 4;
    int    col0 = tn << 4;

    const bf16* arow = X  + (row0 + (size_t)nn) * K + g * 16;  // nn == lane&15 == m
    const bf16* brow = Wt + ((size_t)(col0 + nn)) * K + g * 16;

    v8f acc = {};
    for (int k = 0; k < K; k += 32) {
        v8bf a0 = *(const v8bf*)(arow + k);
        v8bf a1 = *(const v8bf*)(arow + k + 8);
        v16bf a = __builtin_shufflevector(a0, a1, 0,1,2,3,4,5,6,7,8,9,10,11,12,13,14,15);
        v16bf b = *(const v16bf*)(brow + k);
        acc = __builtin_amdgcn_wmma_f32_16x16x32_bf16(false, a, false, b,
                                                      (short)0, acc, false, false);
    }

    float bv = bias ? bias[col0 + nn] : 0.0f;
    #pragma unroll
    for (int r = 0; r < 8; r++) {
        size_t row = row0 + (size_t)(r + g * 8);
        float v = acc[r] + bv;
        if (relu) v = v > 0.f ? v : 0.f;
        size_t idx = row * Nc + col0 + nn;
        if (outF) outF[idx] = v;
        if (outB) outB[idx] = (bf16)v;
    }
}

// ---------------- attention score per (node, head) ----------------
__global__ __launch_bounds__(256) void att_scores_kernel(
    const float* __restrict__ xw, const float* __restrict__ att_src,
    const float* __restrict__ att_dst, float* __restrict__ as_, float* __restrict__ ad_,
    int N, int H, int C)
{
    int t = blockIdx.x * blockDim.x + threadIdx.x;
    if (t >= N * H) return;
    int n = t / H, h = t - n * H;
    const float* xp = xw + (size_t)n * H * C + (size_t)h * C;
    const float* sp = att_src + (size_t)h * C;
    const float* dp = att_dst + (size_t)h * C;
    float a = 0.f, b = 0.f;
    for (int c = 0; c < C; c++) { float v = xp[c]; a += v * sp[c]; b += v * dp[c]; }
    as_[t] = a; ad_[t] = b;
}

// ordered-int encoding so atomicMax(int) implements float max (total order)
__device__ __forceinline__ int f2ord(float f) {
    int b = __float_as_int(f);
    return b >= 0 ? b : (b ^ 0x7fffffff);
}
__device__ __forceinline__ float ord2f(int v) {
    int b = v >= 0 ? v : (v ^ 0x7fffffff);
    return __int_as_float(b);
}

// pass 1: ea = leaky_relu(a_s[src]+a_d[dst]); segment max into mmax[dst]
__global__ __launch_bounds__(256) void edge_logits_kernel(
    const float* __restrict__ as_, const float* __restrict__ ad_,
    const long long* __restrict__ src, const long long* __restrict__ dst,
    float* __restrict__ ea, int* __restrict__ mmax, int E, int H)
{
    int t = blockIdx.x * blockDim.x + threadIdx.x;
    if (t >= E * H) return;
    int e = t / H, h = t - e * H;
    int s = (int)src[e], d = (int)dst[e];
    float v = as_[s * H + h] + ad_[d * H + h];
    v = v > 0.f ? v : NEG_SLOPE * v;
    ea[t] = v;
    atomicMax(&mmax[d * H + h], f2ord(v));
}

// pass 2: ex = exp(ea - m[dst]); segment sum into den[dst] (ea overwritten with ex)
__global__ __launch_bounds__(256) void edge_exp_kernel(
    float* __restrict__ ea, const int* __restrict__ mmax, float* __restrict__ den,
    const long long* __restrict__ dst, int E, int H)
{
    int t = blockIdx.x * blockDim.x + threadIdx.x;
    if (t >= E * H) return;
    int e = t / H, h = t - e * H;
    int d = (int)dst[e];
    float m = ord2f(mmax[d * H + h]);
    if (!isfinite(m)) m = 0.f;                 // empty-segment guard (matches reference)
    float ex = expf(ea[t] - m);
    ea[t] = ex;
    atomicAdd(&den[d * H + h], ex);
}

// pass 3: agg[dst] += xw[src] * alpha ; thread = (edge, 4 channels)
__global__ __launch_bounds__(256) void edge_msg_kernel(
    const float* __restrict__ xw, const float* __restrict__ ex, const float* __restrict__ den,
    const long long* __restrict__ src, const long long* __restrict__ dst,
    float* __restrict__ agg, int E, int H, int C, int DMc)
{
    int t = blockIdx.x * blockDim.x + threadIdx.x;
    if (t >= E * 64) return;
    int e = t >> 6, q = t & 63;
    int s = (int)src[e], d = (int)dst[e];
    int c4 = q << 2;
    int h = c4 / C;
    float al = ex[e * H + h] / (den[d * H + h] + 1e-16f);
    const float4 xv = *(const float4*)(xw + (size_t)s * DMc + c4);
    float* op = agg + (size_t)d * DMc + c4;
    atomicAdd(op + 0, xv.x * al);
    atomicAdd(op + 1, xv.y * al);
    atomicAdd(op + 2, xv.z * al);
    atomicAdd(op + 3, xv.w * al);
}

// ---------------- BatchNorm stats (coalesced column partial sums) ----------------
__global__ void bn_stats_kernel(const float* __restrict__ y, const float* __restrict__ bias,
                                float* __restrict__ sum, float* __restrict__ ssq,
                                int N, int DMc, int rpb)
{
    int c  = threadIdx.x;                      // blockDim.x == DMc
    int r0 = blockIdx.x * rpb;
    int r1 = r0 + rpb; if (r1 > N) r1 = N;
    float bv = bias[c];
    float s = 0.f, q = 0.f;
    for (int r = r0; r < r1; r++) {
        float v = y[(size_t)r * DMc + c] + bv;
        s += v; q += v * v;
    }
    atomicAdd(&sum[c], s);
    atomicAdd(&ssq[c], q);
}

// BN + ELU + residual (in place on x), optional bf16 shadow copy for next GEMM
__global__ __launch_bounds__(256) void bn_elu_res_kernel(
    float* __restrict__ x, const float* __restrict__ agg, const float* __restrict__ bias,
    const float* __restrict__ sum, const float* __restrict__ ssq,
    const float* __restrict__ gamma, const float* __restrict__ beta,
    bf16* __restrict__ xbf, int N, int DMc)
{
    int t = blockIdx.x * blockDim.x + threadIdx.x;
    if (t >= N * DMc) return;
    int c = t & (DMc - 1);
    float invN = 1.0f / (float)N;
    float mu  = sum[c] * invN;
    float var = ssq[c] * invN - mu * mu;
    float v  = agg[t] + bias[c];
    float xh = gamma[c] * (v - mu) * rsqrtf(var + BN_EPS) + beta[c];
    float el = xh > 0.f ? xh : expm1f(xh);
    float nv = x[t] + el;
    x[t] = nv;
    if (xbf) xbf[t] = (bf16)nv;
}

// final tiny matmul: out[N,6] = y1[N,64] @ W2[64,6] + b2
__global__ __launch_bounds__(256) void mlp_final_kernel(
    const float* __restrict__ y, const float* __restrict__ W, const float* __restrict__ b,
    float* __restrict__ out, int N)
{
    int t = blockIdx.x * blockDim.x + threadIdx.x;
    if (t >= N * 6) return;
    int n = t / 6, c = t - n * 6;
    const float* yp = y + (size_t)n * 64;
    float s = b[c];
    #pragma unroll
    for (int k = 0; k < 64; k++) s += yp[k] * W[k * 6 + c];
    out[t] = s;
}

// ---------------- host orchestration ----------------

static inline int gdiv(long long n, int b) { return (int)((n + b - 1) / b); }

extern "C" void kernel_launch(void* const* d_in, const int* in_sizes, int n_in,
                              void* d_out, int out_size, void* d_ws, size_t ws_size,
                              hipStream_t stream) {
    (void)n_in; (void)out_size; (void)ws_size;
    const float*      h_in   = (const float*)d_in[0];
    const long long*  eidx   = (const long long*)d_in[1];   // int64 [2,E]
    const float*      emb_W  = (const float*)d_in[3];
    const float*      emb_b  = (const float*)d_in[4];
    const float*      gat_W  = (const float*)d_in[5];
    const float*      gat_as = (const float*)d_in[6];
    const float*      gat_ad = (const float*)d_in[7];
    const float*      gat_b  = (const float*)d_in[8];
    const float*      bn_g   = (const float*)d_in[9];
    const float*      bn_b   = (const float*)d_in[10];
    const float*      gatL_W = (const float*)d_in[11];
    const float*      gatL_as= (const float*)d_in[12];
    const float*      gatL_ad= (const float*)d_in[13];
    const float*      gatL_b = (const float*)d_in[14];
    const float*      bnL_g  = (const float*)d_in[15];
    const float*      bnL_b  = (const float*)d_in[16];
    const float*      mlp_W0 = (const float*)d_in[17];
    const float*      mlp_b0 = (const float*)d_in[18];
    const float*      mlp_W1 = (const float*)d_in[19];
    const float*      mlp_b1 = (const float*)d_in[20];
    const float*      mlp_W2 = (const float*)d_in[21];
    const float*      mlp_b2 = (const float*)d_in[22];

    const int N  = in_sizes[0] / 128;
    const int E  = in_sizes[1] / 2;
    const int DM = 256, H = 8, C = 32;
    const long long src_off = 0, dst_off = E;

    // workspace carving (256B aligned)
    char* ws = (char*)d_ws; size_t off = 0;
    auto alloc = [&](size_t bytes) -> char* {
        char* p = ws + off; off += (bytes + 255) & ~(size_t)255; return p;
    };
    float* x    = (float*)alloc((size_t)N * DM * 4);
    float* xw   = (float*)alloc((size_t)N * DM * 4);
    float* agg  = (float*)alloc((size_t)N * DM * 4);
    bf16*  xbf  = (bf16*) alloc((size_t)N * DM * 2);
    float* ea   = (float*)alloc((size_t)E * H * 4);   // reused as hbf / y0bf (same size)
    float* as_  = (float*)alloc((size_t)N * H * 4);
    float* ad_  = (float*)alloc((size_t)N * H * 4);
    int*   mmax = (int*)  alloc((size_t)N * H * 4);
    float* den  = (float*)alloc((size_t)N * H * 4);
    bf16*  embWt = (bf16*)alloc((size_t)256 * 128 * 2);
    bf16*  gatWt = (bf16*)alloc((size_t)3 * 256 * 256 * 2);
    bf16*  gatLWt= (bf16*)alloc((size_t)256 * 256 * 2);
    bf16*  m0Wt  = (bf16*)alloc((size_t)128 * 256 * 2);
    bf16*  m1Wt  = (bf16*)alloc((size_t)64 * 128 * 2);
    float* bsum  = (float*)alloc(256 * 4);
    float* bsq   = (float*)alloc(256 * 4);

    // weights -> bf16, transposed [Nout][K]
    transpose_to_bf16_kernel<<<gdiv(128 * 256, 256), 256, 0, stream>>>(emb_W, embWt, 128, 256);
    for (int i = 0; i < 3; i++)
        transpose_to_bf16_kernel<<<gdiv(256 * 256, 256), 256, 0, stream>>>(
            gat_W + (size_t)i * 256 * 256, gatWt + (size_t)i * 256 * 256, 256, 256);
    transpose_to_bf16_kernel<<<gdiv(256 * 256, 256), 256, 0, stream>>>(gatL_W, gatLWt, 256, 256);
    transpose_to_bf16_kernel<<<gdiv(256 * 128, 256), 256, 0, stream>>>(mlp_W0, m0Wt, 256, 128);
    transpose_to_bf16_kernel<<<gdiv(128 * 64, 256), 256, 0, stream>>>(mlp_W1, m1Wt, 128, 64);

    // input embedding: x = h @ emb_W + emb_b  (also emit bf16 shadow)
    bf16* hbf = (bf16*)ea;
    f32_to_bf16_kernel<<<gdiv((long long)N * 128, 256), 256, 0, stream>>>(h_in, hbf, N * 128);
    {
        int tiles = (N >> 4) * (DM >> 4);
        wmma_gemm_kernel<<<gdiv(tiles, 8), 256, 0, stream>>>(
            hbf, embWt, emb_b, x, xbf, N, 128, DM, 0);
    }

    // one GAT layer: xw = x@W ; attention softmax over incoming edges ; BN+ELU+residual
    auto gat_layer = [&](const bf16* Wt, const float* a_s, const float* a_d,
                         const float* bias, const float* g, const float* bvec,
                         int Hh, int Cc, bf16* xbf_out) {
        int tiles = (N >> 4) * (DM >> 4);
        wmma_gemm_kernel<<<gdiv(tiles, 8), 256, 0, stream>>>(
            xbf, Wt, nullptr, xw, nullptr, N, DM, DM, 0);
        att_scores_kernel<<<gdiv((long long)N * Hh, 256), 256, 0, stream>>>(
            xw, a_s, a_d, as_, ad_, N, Hh, Cc);
        fill_i32_kernel<<<gdiv((long long)N * Hh, 256), 256, 0, stream>>>(mmax, (int)0x80000000, N * Hh);
        fill_f32_kernel<<<gdiv((long long)N * Hh, 256), 256, 0, stream>>>(den, 0.f, N * Hh);
        fill_f32_kernel<<<gdiv((long long)N * DM, 256), 256, 0, stream>>>(agg, 0.f, N * DM);
        fill_f32_kernel<<<1, 256, 0, stream>>>(bsum, 0.f, 256);
        fill_f32_kernel<<<1, 256, 0, stream>>>(bsq, 0.f, 256);
        edge_logits_kernel<<<gdiv((long long)E * Hh, 256), 256, 0, stream>>>(
            as_, ad_, eidx + src_off, eidx + dst_off, ea, mmax, E, Hh);
        edge_exp_kernel<<<gdiv((long long)E * Hh, 256), 256, 0, stream>>>(
            ea, mmax, den, eidx + dst_off, E, Hh);
        edge_msg_kernel<<<gdiv((long long)E * 64, 256), 256, 0, stream>>>(
            xw, ea, den, eidx + src_off, eidx + dst_off, agg, E, Hh, Cc, DM);
        bn_stats_kernel<<<gdiv(N, 128), DM, 0, stream>>>(agg, bias, bsum, bsq, N, DM, 128);
        bn_elu_res_kernel<<<gdiv((long long)N * DM, 256), 256, 0, stream>>>(
            x, agg, bias, bsum, bsq, g, bvec, xbf_out, N, DM);
    };

    for (int i = 0; i < 3; i++)
        gat_layer(gatWt + (size_t)i * 256 * 256,
                  gat_as + (size_t)i * H * C, gat_ad + (size_t)i * H * C,
                  gat_b + (size_t)i * DM, bn_g + (size_t)i * DM, bn_b + (size_t)i * DM,
                  H, C, xbf);
    gat_layer(gatLWt, gatL_as, gatL_ad, gatL_b, bnL_g, bnL_b, 1, 256, xbf);

    // MLP readout: 256 -> 128 (relu, bf16) -> 64 (relu, f32) -> 6
    bf16* y0 = (bf16*)ea;       // ea region free now
    {
        int tiles = (N >> 4) * (128 >> 4);
        wmma_gemm_kernel<<<gdiv(tiles, 8), 256, 0, stream>>>(
            xbf, m0Wt, mlp_b0, nullptr, y0, N, 256, 128, 1);
    }
    float* y1 = agg;            // agg region free now
    {
        int tiles = (N >> 4) * (64 >> 4);
        wmma_gemm_kernel<<<gdiv(tiles, 8), 256, 0, stream>>>(
            y0, m1Wt, mlp_b1, y1, nullptr, N, 128, 64, 1);
    }
    mlp_final_kernel<<<gdiv((long long)N * 6, 256), 256, 0, stream>>>(
        y1, mlp_W2, mlp_b2, (float*)d_out, N);
}
